// MixtureOfClassifiers_24103356465355
// MI455X (gfx1250) — matrix operand order repaired and checked by reading, not compiled, not run on hardware
//
#include <hip/hip_runtime.h>
#include <math.h>
#include <stdint.h>

// ---------------------------------------------------------------------------
// MixtureOfClassifiers on MI455X (gfx1250, wave32, WMMA bf16 16x16x32)
//   K0: pre-pack W1|We fp32 -> bf16 Bpack[D/32][384][32] (LDS tile layout)
//   K1: fused dual GEMM over x (read x ONCE): [B,4096] x [4096, 256+128]
//       double-buffered LDS, async global->LDS copies, pipelined B fragments
//       -> hidden bf16 [B,256] (relu(+b1)), expert_out f32 [B,128] (+be)
//   K2: router 2nd layer + gumbel softmax -> weights f32 [B,64]
//   K3: weighted mixture -> out [B,2]
// ---------------------------------------------------------------------------

#define B_ROWS 16384
#define DDIM   4096
#define HDIM   256
#define EDIM   64
#define NOUT   128          // E*2
#define NTOT   384          // HDIM + NOUT
#define KSTEP  32

typedef __attribute__((ext_vector_type(16))) __bf16 v16bf;
typedef __attribute__((ext_vector_type(8)))  __bf16 v8bf;
typedef __attribute__((ext_vector_type(8)))  float  v8f;
typedef __attribute__((ext_vector_type(4)))  int    v4i;

#if defined(__has_builtin)
#if __has_builtin(__builtin_amdgcn_global_load_async_to_lds_b128) && \
    __has_builtin(__builtin_amdgcn_s_wait_asynccnt)
#define USE_ASYNC_LDS 1
#endif
#endif

__device__ __forceinline__ unsigned short f2bfu(float f) {
    union { float f; unsigned u; } v; v.f = f;
    unsigned r = (v.u + 0x7FFFu + ((v.u >> 16) & 1u)) >> 16;   // RNE
    return (unsigned short)r;
}
__device__ __forceinline__ float bfu2f(unsigned short s) {
    union { unsigned u; float f; } v; v.u = ((unsigned)s) << 16;
    return v.f;
}
__device__ __forceinline__ unsigned packbf(float lo, float hi) {
    return (unsigned)f2bfu(lo) | ((unsigned)f2bfu(hi) << 16);
}

// Async global->LDS b128 copy (ASYNCcnt path); fallback = regular load+store.
__device__ __forceinline__ void copy_b128_to_lds(const uint4* __restrict__ g,
                                                 uint4* l) {
#ifdef USE_ASYNC_LDS
    __builtin_amdgcn_global_load_async_to_lds_b128(
        (__attribute__((address_space(1))) v4i*)(uintptr_t)g,
        (__attribute__((address_space(3))) v4i*)(unsigned)(uintptr_t)l,
        0, 0);
#else
    *l = *g;
#endif
}
__device__ __forceinline__ void wait_async_lds() {
#ifdef USE_ASYNC_LDS
    __builtin_amdgcn_s_wait_asynccnt(0);
#endif
}

// ---------------------------------------------------------------------------
// K0: pack W1 (cols 0..255) and We (cols 256..383) into bf16, tile layout
//     Bpack[(d/32)][n][d%32]
// ---------------------------------------------------------------------------
__global__ __launch_bounds__(256)
void moc_pack_k0(const float* __restrict__ W1,
                 const float* __restrict__ We,
                 unsigned short* __restrict__ Bpack)
{
    const int idx = blockIdx.x * 256 + threadIdx.x;     // < DDIM*NTOT
    const int kc  = idx / (NTOT * KSTEP);
    const int rem = idx - kc * (NTOT * KSTEP);
    const int n   = rem / KSTEP;
    const int k   = rem - n * KSTEP;
    const int d   = kc * KSTEP + k;
    float g;
    if (n < HDIM) {
        g = W1[(size_t)d * HDIM + n];
    } else {
        const int c = n - HDIM;                         // c = e*2 + o
        g = We[((size_t)(c >> 1) * DDIM + d) * 2 + (c & 1)];
    }
    Bpack[idx] = f2bfu(g);
}

// ---------------------------------------------------------------------------
// K1: dual GEMM.  grid = 256 blocks (64 rows each), block = 256 threads (8 waves)
// wave layout: wm = w&3 (row tile of 16), wn = w>>2 (192-col half) -> 12 N-tiles
// ---------------------------------------------------------------------------
__global__ __launch_bounds__(256)
void moc_dualgemm_k1(const float* __restrict__ x,
                     const unsigned short* __restrict__ Bpack,
                     const float* __restrict__ b1,
                     const float* __restrict__ be,
                     unsigned short* __restrict__ hidden,   // bf16 bits [B,256]
                     float* __restrict__ eo)                // [B,128]
{
    __shared__ __align__(16) unsigned short Ash[2][64 * KSTEP];    //  8 KB
    __shared__ __align__(16) unsigned short Bsh[2][NTOT * KSTEP];  // 48 KB

    const int tid  = threadIdx.x;
    const int w    = tid >> 5;
    const int lane = tid & 31;
    const int wm   = w & 3;          // 0..3  -> row tile
    const int wn   = w >> 2;         // 0..1  -> col half
    const int m0g  = blockIdx.x * 64;
    const int nbase = wn * 192;
    const int lm = lane & 15;
    const int lh = lane >> 4;

    // staging helpers -------------------------------------------------------
    const int arow = tid >> 2;
    const int akg  = (tid & 3) * 8;

    auto stageA = [&](int k0, int buf) {
        const float* gp = x + (size_t)(m0g + arow) * DDIM + k0 + akg;
        float4 f0 = *(const float4*)gp;
        float4 f1 = *(const float4*)(gp + 4);
        uint4 p;
        p.x = packbf(f0.x, f0.y); p.y = packbf(f0.z, f0.w);
        p.z = packbf(f1.x, f1.y); p.w = packbf(f1.z, f1.w);
        *(uint4*)&Ash[buf][arow * KSTEP + akg] = p;
    };
    auto stageB = [&](int k0, int buf) {
        const uint4* src = (const uint4*)(Bpack + (size_t)(k0 / KSTEP) * NTOT * KSTEP);
        uint4* dst = (uint4*)&Bsh[buf][0];
        #pragma unroll
        for (int j = 0; j < (NTOT * KSTEP * 2 / 16) / 256; ++j)   // 6 x b128/thread
            copy_b128_to_lds(src + tid + j * 256, dst + tid + j * 256);
    };
    // B fragment loader: col = nbase + t*16 + lm, K runs {lh*8..+7, 16+lh*8..+7}
    auto loadBfrag = [&](int buf, int t) -> v16bf {
        const unsigned short* bp =
            &Bsh[buf][(nbase + t * 16 + lm) * KSTEP];
        v8bf blo = *(const v8bf*)(bp + lh * 8);
        v8bf bhi = *(const v8bf*)(bp + 16 + lh * 8);
        v16bf bb;
        #pragma unroll
        for (int j = 0; j < 8; ++j) { bb[j] = blo[j]; bb[j + 8] = bhi[j]; }
        return bb;
    };

    v8f acc[12];
    #pragma unroll
    for (int t = 0; t < 12; ++t) acc[t] = {};

    // prologue: fill buffer 0
    stageB(0, 0);
    stageA(0, 0);
    wait_async_lds();
    __syncthreads();

    for (int k0 = 0; k0 < DDIM; k0 += KSTEP) {
        const int buf = (k0 / KSTEP) & 1;
        // prefetch next chunk into the other buffer (overlaps with WMMA)
        if (k0 + KSTEP < DDIM) {
            stageB(k0 + KSTEP, buf ^ 1);
            stageA(k0 + KSTEP, buf ^ 1);
        }

        // ---- A fragment: row = wm*16 + lm, K runs {lh*8..+7, 16+lh*8..+7} --
        v16bf a;
        {
            const unsigned short* ap = &Ash[buf][(wm * 16 + lm) * KSTEP];
            v8bf alo = *(const v8bf*)(ap + lh * 8);
            v8bf ahi = *(const v8bf*)(ap + 16 + lh * 8);
            #pragma unroll
            for (int j = 0; j < 8; ++j) { a[j] = alo[j]; a[j + 8] = ahi[j]; }
        }
        // ---- 12 WMMAs, software-pipelined B fragments (hide LDS latency) ---
        v16bf bcur = loadBfrag(buf, 0);
        #pragma unroll
        for (int t = 0; t < 12; ++t) {
            v16bf bnxt;
            if (t < 11) bnxt = loadBfrag(buf, t + 1);
            acc[t] = __builtin_amdgcn_wmma_f32_16x16x32_bf16(
                false, a, false, bcur, (short)0, acc[t], false, false);
            if (t < 11) bcur = bnxt;
        }

        wait_async_lds();     // next-chunk async copies landed
        __syncthreads();      // everyone done with buf + next buffer ready
    }

    // ---- epilogue: C layout M = r + lh*8, N = lm ---------------------------
    #pragma unroll
    for (int t = 0; t < 12; ++t) {
        const int col = nbase + t * 16 + lm;
        #pragma unroll
        for (int r = 0; r < 8; ++r) {
            const int row = m0g + wm * 16 + r + lh * 8;
            float v = acc[t][r];
            if (col < HDIM) {
                v += b1[col];
                v = fmaxf(v, 0.0f);
                hidden[(size_t)row * HDIM + col] = f2bfu(v);
            } else {
                const int c = col - HDIM;
                v += be[c];
                eo[(size_t)row * NOUT + c] = v;
            }
        }
    }
}

// ---------------------------------------------------------------------------
// K2: logits = hidden @ W2 + b2; gumbel; softmax -> weights [B,64]
// grid = 2048 blocks, block = 256 threads = 8 waves, one row per wave
// ---------------------------------------------------------------------------
__global__ __launch_bounds__(256)
void moc_router_k2(const unsigned short* __restrict__ hidden,
                   const float* __restrict__ W2,
                   const float* __restrict__ b2,
                   const float* __restrict__ u,
                   float* __restrict__ wts)
{
    __shared__ unsigned short Hsh[8 * HDIM];     // 4 KB
    const int tid  = threadIdx.x;
    const int w    = tid >> 5;
    const int lane = tid & 31;
    const int rowbase = blockIdx.x * 8;

    #pragma unroll
    for (int j = 0; j < (8 * HDIM) / 256; ++j) {
        const int i = tid + j * 256;
        Hsh[i] = hidden[(size_t)(rowbase + (i >> 8)) * HDIM + (i & 255)];
    }
    __syncthreads();

    const int row = rowbase + w;
    float acc0 = b2[lane];
    float acc1 = b2[lane + 32];
    #pragma unroll 8
    for (int h = 0; h < HDIM; ++h) {
        const float xh = bfu2f(Hsh[w * HDIM + h]);
        acc0 = fmaf(xh, W2[h * EDIM + lane],      acc0);
        acc1 = fmaf(xh, W2[h * EDIM + lane + 32], acc1);
    }
    // gumbel noise (TEMPERATURE = 1)
    const float u0 = u[(size_t)row * EDIM + lane];
    const float u1 = u[(size_t)row * EDIM + lane + 32];
    float l0 = acc0 - logf(-logf(u0 + 1e-8f) + 1e-8f);
    float l1 = acc1 - logf(-logf(u1 + 1e-8f) + 1e-8f);
    // wave-wide softmax over 64 values (2 per lane)
    float m = fmaxf(l0, l1);
    #pragma unroll
    for (int off = 16; off >= 1; off >>= 1) m = fmaxf(m, __shfl_xor(m, off, 32));
    const float e0 = expf(l0 - m), e1 = expf(l1 - m);
    float s = e0 + e1;
    #pragma unroll
    for (int off = 16; off >= 1; off >>= 1) s += __shfl_xor(s, off, 32);
    const float inv = 1.0f / s;
    wts[(size_t)row * EDIM + lane]      = e0 * inv;
    wts[(size_t)row * EDIM + lane + 32] = e1 * inv;
}

// ---------------------------------------------------------------------------
// K3: out[b,o] = sum_e wts[b,e] * eo[b, e*2+o].  One wave per row.
// ---------------------------------------------------------------------------
__global__ __launch_bounds__(256)
void moc_mix_k3(const float* __restrict__ eo,
                const float* __restrict__ wts,
                float* __restrict__ out)
{
    const int tid  = threadIdx.x;
    const int w    = tid >> 5;
    const int lane = tid & 31;
    const int row  = blockIdx.x * 8 + w;

    const float w0 = wts[(size_t)row * EDIM + lane];
    const float w1 = wts[(size_t)row * EDIM + lane + 32];
    const float* er = eo + (size_t)row * NOUT;
    float a0 = w0 * er[lane * 2]     + w1 * er[(lane + 32) * 2];
    float a1 = w0 * er[lane * 2 + 1] + w1 * er[(lane + 32) * 2 + 1];
    #pragma unroll
    for (int off = 16; off >= 1; off >>= 1) {
        a0 += __shfl_xor(a0, off, 32);
        a1 += __shfl_xor(a1, off, 32);
    }
    if (lane == 0) {
        out[(size_t)row * 2]     = a0;
        out[(size_t)row * 2 + 1] = a1;
    }
}

// ---------------------------------------------------------------------------
extern "C" void kernel_launch(void* const* d_in, const int* in_sizes, int n_in,
                              void* d_out, int out_size, void* d_ws, size_t ws_size,
                              hipStream_t stream)
{
    const float* x  = (const float*)d_in[0];
    const float* u  = (const float*)d_in[1];
    const float* W1 = (const float*)d_in[2];
    const float* b1 = (const float*)d_in[3];
    const float* W2 = (const float*)d_in[4];
    const float* b2 = (const float*)d_in[5];
    const float* We = (const float*)d_in[6];
    const float* be = (const float*)d_in[7];
    float* out = (float*)d_out;

    char* ws = (char*)d_ws;
    unsigned short* hidden = (unsigned short*)ws;                         // 8 MB bf16
    float* eo   = (float*)(ws + (size_t)B_ROWS * HDIM * 2);               // 8 MB
    float* wts  = (float*)(ws + (size_t)B_ROWS * HDIM * 2
                              + (size_t)B_ROWS * NOUT * 4);               // 4 MB
    unsigned short* Bpack = (unsigned short*)(ws + (size_t)B_ROWS * HDIM * 2
                                                 + (size_t)B_ROWS * NOUT * 4
                                                 + (size_t)B_ROWS * EDIM * 4); // 3 MB

    moc_pack_k0   <<<(DDIM * NTOT) / 256, 256, 0, stream>>>(W1, We, Bpack);
    moc_dualgemm_k1<<<B_ROWS / 64, 256, 0, stream>>>(x, Bpack, b1, be, hidden, eo);
    moc_router_k2 <<<B_ROWS / 8,  256, 0, stream>>>(hidden, W2, b2, u, wts);
    moc_mix_k3    <<<B_ROWS / 8,  256, 0, stream>>>(eo, wts, out);
}
